// MultiHeadSelfAttention_58394375357204
// MI455X (gfx1250) — matrix operand order repaired and checked by reading, not compiled, not run on hardware
//
#include <hip/hip_runtime.h>
#include <hip/hip_bf16.h>

typedef _Float16 v4h  __attribute__((ext_vector_type(4)));
typedef _Float16 v8h  __attribute__((ext_vector_type(8)));
typedef _Float16 v16h __attribute__((ext_vector_type(16)));
typedef float    v8f  __attribute__((ext_vector_type(8)));
typedef float    v4f  __attribute__((ext_vector_type(4)));
typedef int      i32x4 __attribute__((vector_size(16)));

#define DM   1024
#define NH   16
#define DH   64
#define SEQ  2048
#define NB   4
#define MTOT (NB*SEQ)       // 8192 rows
#define LOG2E 1.44269504089f
#define LDSP 72             // attention LDS row stride (halfs): conflict-free b128 reads
#define GSP  40             // gemm LDS row stride (halfs): 32 data + 8 pad, conflict-free

static __device__ __forceinline__ v8f wmma_f16(v16h a, v16h b, v8f c) {
  return __builtin_amdgcn_wmma_f32_16x16x32_f16(false, a, false, b, (short)0, c, false, false);
}

// B-fragment: 16 contiguous halfs at p (caller adds hi*16). element i <-> K = hi*16 + i
static __device__ __forceinline__ v16h ldB16(const _Float16* p) {
  v8h lo = *(const v8h*)p;
  v8h hi = *(const v8h*)(p + 8);
  return __builtin_shufflevector(lo, hi, 0,1,2,3,4,5,6,7,8,9,10,11,12,13,14,15);
}

// A-fragment: p includes +hi*8; elems 0-7 at K=hi*8.., elems 8-15 at K=16+hi*8..
static __device__ __forceinline__ v16h ldA16(const _Float16* p) {
  v8h lo = *(const v8h*)p;
  v8h hi = *(const v8h*)(p + 16);
  return __builtin_shufflevector(lo, hi, 0,1,2,3,4,5,6,7,8,9,10,11,12,13,14,15);
}

// ---- async global->LDS staging (gfx1250 ASYNC path), sync fallback if builtin absent ----
#if __has_builtin(__builtin_amdgcn_global_load_async_to_lds_b128)
#define HAVE_ASYNC_LDS 1
typedef __attribute__((address_space(1))) i32x4 gas_i32x4;
typedef __attribute__((address_space(3))) i32x4 las_i32x4;
static __device__ __forceinline__ void cp_b128(_Float16* dst, const _Float16* src) {
  __builtin_amdgcn_global_load_async_to_lds_b128(
      (gas_i32x4*)src, (las_i32x4*)dst, 0, 0);
}
#else
#define HAVE_ASYNC_LDS 0
static __device__ __forceinline__ void cp_b128(_Float16* dst, const _Float16* src) {
  *(v8h*)dst = *(const v8h*)src;
}
#endif

#if __has_builtin(__builtin_amdgcn_s_wait_asynccnt)
#define WAIT_ASYNC(n) __builtin_amdgcn_s_wait_asynccnt(n)
#elif HAVE_ASYNC_LDS
#define WAIT_ASYNC(n) asm volatile("s_wait_asynccnt %0" ::"n"(n))
#else
#define WAIT_ASYNC(n) ((void)0)
#endif

// ---------------- f32 -> f16 conversion (one pass, amortized) ----------------
__global__ __launch_bounds__(256) void k_cvt_f32_f16(const float* __restrict__ src,
                                                     _Float16* __restrict__ dst, int n) {
  int i = (blockIdx.x * 256 + threadIdx.x) * 4;
  if (i < n) {
    v4f v = *(const v4f*)(src + i);
    v4h h;
#pragma unroll
    for (int t = 0; t < 4; ++t) h[t] = (_Float16)v[t];
    *(v4h*)(dst + i) = h;
  }
}

// ---------------- shared GEMM mainloop: 128Mx128N macro-tile, async-LDS double buffer ----
// 8 waves = 4 m-waves x 2 n-waves; wave tile 32Mx64N (acc[2][4]); K staged in 32-chunks.
static __device__ __forceinline__ void gemm_mainloop_128x128(
    const _Float16* __restrict__ Ab,   // global A panel base (128 rows x DM)
    const _Float16* __restrict__ Bb,   // global B panel base (128 rows x DM)
    _Float16* smA, _Float16* smB,      // [2][128*GSP] each
    v8f (&acc)[2][4], int tid) {
  const int lane = tid & 31;
  const int wid  = tid >> 5;
  const int mw = wid & 3, nw = wid >> 2;
  const int row = lane & 15, hi = lane >> 4;
  // staging: 512 b128 chunks per 128x32 panel, 2 per thread per panel (coalesced rows)
  const int sr0 = tid >> 2,          sc0 = (tid & 3) * 8;
  const int sr1 = (tid + 256) >> 2;  // rows 64..127

#define GSTAGE(buf, kb)                                                           \
  {                                                                               \
    cp_b128(smA + (buf)*128*GSP + sr0*GSP + sc0, Ab + sr0*DM + (kb) + sc0);       \
    cp_b128(smA + (buf)*128*GSP + sr1*GSP + sc0, Ab + sr1*DM + (kb) + sc0);       \
    cp_b128(smB + (buf)*128*GSP + sr0*GSP + sc0, Bb + sr0*DM + (kb) + sc0);       \
    cp_b128(smB + (buf)*128*GSP + sr1*GSP + sc0, Bb + sr1*DM + (kb) + sc0);       \
  }

  const int NS = DM / 32;
  GSTAGE(0, 0);
  for (int s = 0; s < NS; ++s) {
    const int cur = s & 1;
    if (s + 1 < NS) { GSTAGE(cur ^ 1, (s + 1) * 32); WAIT_ASYNC(4); }
    else            { WAIT_ASYNC(0); }
    __syncthreads();

    const _Float16* As = smA + cur * 128 * GSP + (mw * 32) * GSP;
    const _Float16* Bs = smB + cur * 128 * GSP + (nw * 64) * GSP;
    v16h af0 = ldA16(As + row * GSP + hi * 8);
    v16h af1 = ldA16(As + (16 + row) * GSP + hi * 8);
#pragma unroll
    for (int j = 0; j < 4; ++j) {
      v16h bf = ldA16(Bs + (j * 16 + row) * GSP + hi * 8);
      acc[0][j] = wmma_f16(af0, bf, acc[0][j]);
      acc[1][j] = wmma_f16(af1, bf, acc[1][j]);
    }
    __syncthreads();
  }
#undef GSTAGE
}

// ---------------- QKV GEMM: [8192,1024] x [3072,1024]^T -> Q/K [B,H,T,64], V^T [B,H,64,T]
__global__ __launch_bounds__(256) void k_gemm_qkv(const _Float16* __restrict__ Xh,
                                                  const _Float16* __restrict__ Wh,
                                                  _Float16* __restrict__ Qd,
                                                  _Float16* __restrict__ Kd,
                                                  _Float16* __restrict__ Vt) {
  __shared__ _Float16 smA[2][128 * GSP];
  __shared__ _Float16 smB[2][128 * GSP];
  const int tid  = threadIdx.x;
  const int lane = tid & 31;
  const int wid  = tid >> 5;
  const int nblk = blockIdx.x % 24;             // 128-wide N block
  const int mblk = blockIdx.x / 24;             // 128-wide M block
  const int mw = wid & 3, nw = wid >> 2;
  const int row = lane & 15, hi = lane >> 4;

  v8f acc[2][4] = {};
  gemm_mainloop_128x128(Xh + (size_t)mblk * 128 * DM, Wh + (size_t)nblk * 128 * DM,
                        &smA[0][0], &smB[0][0], acc, tid);

  const int nbase = nblk * 128 + nw * 64;       // 64-aligned
  const int part  = nbase >> 10;                // 0=Q 1=K 2=V (constant per wave)
  const int h     = (nbase & 1023) >> 6;
  const int mbase = mblk * 128 + mw * 32;
  const int bb    = mbase >> 11;

#pragma unroll
  for (int a = 0; a < 2; ++a) {
    const int t0 = (mbase + a * 16) & 2047;
    if (part < 2) {
      _Float16* dst = (part == 0) ? Qd : Kd;
      const float scale = (part == 0) ? 0.125f : 1.0f;   // 1/sqrt(64) folded into Q
#pragma unroll
      for (int j = 0; j < 4; ++j) {
        const int dh = j * 16 + row;
#pragma unroll
        for (int r = 0; r < 8; ++r) {
          int t = t0 + r + hi * 8;
          dst[(((bb * NH + h) * SEQ) + t) * DH + dh] = (_Float16)(acc[a][j][r] * scale);
        }
      }
    } else {
      // V transposed: lane owns 8 consecutive t -> one b128 store per j-tile
#pragma unroll
      for (int j = 0; j < 4; ++j) {
        const int dh = j * 16 + row;
        v8h col;
#pragma unroll
        for (int r = 0; r < 8; ++r) col[r] = (_Float16)acc[a][j][r];
        *(v8h*)(Vt + ((size_t)((bb * NH + h) * DH + dh)) * SEQ + t0 + hi * 8) = col;
      }
    }
  }
}

// ---------------- Flash attention, block-cooperative K/V staging in LDS ----------------
__global__ __launch_bounds__(256) void k_attn(const _Float16* __restrict__ Qd,
                                              const _Float16* __restrict__ Kd,
                                              const _Float16* __restrict__ Vt,
                                              _Float16* __restrict__ Od) {
  __shared__ _Float16 smK[2][64 * LDSP];
  __shared__ _Float16 smV[2][64 * LDSP];

  const int tid  = threadIdx.x;
  const int lane = tid & 31;
  const int wid  = tid >> 5;
  const int bh   = blockIdx.x >> 4;             // (b*16+h)
  const int qblk = blockIdx.x & 15;             // 128-query block
  const int qt   = qblk * 8 + wid;              // this wave's 16-query tile
  const int row = lane & 15, hi = lane >> 4;

  const _Float16* Qb  = Qd + bh * SEQ * DH;
  const _Float16* Kb  = Kd + bh * SEQ * DH;
  const _Float16* Vtb = Vt + bh * DH * SEQ;     // [dh][t]

  const _Float16* pq = Qb + (qt * 16 + row) * DH + hi * 8;
  const v16h bq0 = ldA16(pq);        // dh 0..31
  const v16h bq1 = ldA16(pq + 32);   // dh 32..63

  const int r0 = tid >> 3, c0 = (tid & 7) * 8;
  const int r1 = (tid + 256) >> 3;
#define STAGE(buf, kb)                                                              \
  {                                                                                 \
    cp_b128(&smK[buf][r0 * LDSP + c0], Kb + ((kb) + r0) * DH + c0);                 \
    cp_b128(&smK[buf][r1 * LDSP + c0], Kb + ((kb) + r1) * DH + c0);                 \
    cp_b128(&smV[buf][r0 * LDSP + c0], Vtb + (size_t)r0 * SEQ + (kb) + c0);         \
    cp_b128(&smV[buf][r1 * LDSP + c0], Vtb + (size_t)r1 * SEQ + (kb) + c0);         \
  }

  float mrun = -3.0e38f, lrun = 0.0f;
  v8f acc[4] = {};
  const v8f zero = {};
  const int NSTEP = SEQ / 64;

  STAGE(0, 0);

  for (int s = 0; s < NSTEP; ++s) {
    const int cur = s & 1;
    if (s + 1 < NSTEP) { STAGE(cur ^ 1, (s + 1) * 64); WAIT_ASYNC(4); }
    else               { WAIT_ASYNC(0); }
    __syncthreads();

    const _Float16* Ks = &smK[cur][0];
    const _Float16* Vs = &smV[cur][0];

    v8f st[4];
#pragma unroll
    for (int g = 0; g < 4; ++g) {
      const _Float16* pk = Ks + (g * 16 + row) * LDSP + hi * 8;
      st[g] = wmma_f16(ldA16(pk), bq0, zero);
      st[g] = wmma_f16(ldA16(pk + 32), bq1, st[g]);
    }

    float tm = -3.0e38f;
#pragma unroll
    for (int g = 0; g < 4; ++g)
#pragma unroll
      for (int r = 0; r < 8; ++r) tm = fmaxf(tm, st[g][r]);
    tm = fmaxf(tm, __shfl_xor(tm, 16, 32));
    const float nm = fmaxf(mrun, tm);
    const float alpha = exp2f((mrun - nm) * LOG2E);
    float p[4][8], ls = 0.0f;
#pragma unroll
    for (int g = 0; g < 4; ++g)
#pragma unroll
      for (int r = 0; r < 8; ++r) { p[g][r] = exp2f((st[g][r] - nm) * LOG2E); ls += p[g][r]; }
    ls += __shfl_xor(ls, 16, 32);
    lrun = lrun * alpha + ls;
    mrun = nm;

    float ar[8];
#pragma unroll
    for (int r = 0; r < 8; ++r) ar[r] = __shfl(alpha, r + 8 * hi, 32);
#pragma unroll
    for (int j = 0; j < 4; ++j)
#pragma unroll
      for (int r = 0; r < 8; ++r) acc[j][r] *= ar[r];

    v16h ap0, ap1;
#pragma unroll
    for (int r = 0; r < 8; ++r) {
      ap0[r] = (_Float16)p[0][r]; ap0[8 + r] = (_Float16)p[1][r];
      ap1[r] = (_Float16)p[2][r]; ap1[8 + r] = (_Float16)p[3][r];
    }

#pragma unroll
    for (int j = 0; j < 4; ++j) {
      const _Float16* pv = Vs + (j * 16 + row) * LDSP + hi * 8;
      acc[j] = wmma_f16(ap0, ldA16(pv), acc[j]);
      acc[j] = wmma_f16(ap1, ldA16(pv + 32), acc[j]);
    }
    __syncthreads();
  }
#undef STAGE

  const float li = 1.0f / lrun;
  float lr[8];
#pragma unroll
  for (int r = 0; r < 8; ++r) lr[r] = __shfl(li, r + 8 * hi, 32);

  const int bb = bh >> 4, h = bh & 15;
#pragma unroll
  for (int j = 0; j < 4; ++j) {
    const int dh = h * DH + j * 16 + row;
#pragma unroll
    for (int r = 0; r < 8; ++r) {
      int t = qt * 16 + r + hi * 8;
      Od[((size_t)(bb * SEQ + t)) * DM + dh] = (_Float16)(acc[j][r] * lr[r]);
    }
  }
}

// ---------------- Output projection: [8192,1024] x [1024,1024]^T -> f32 out ----------------
__global__ __launch_bounds__(256) void k_gemm_proj(const _Float16* __restrict__ Ah,
                                                   const _Float16* __restrict__ Wh,
                                                   float* __restrict__ out) {
  __shared__ _Float16 smA[2][128 * GSP];
  __shared__ _Float16 smB[2][128 * GSP];
  const int tid  = threadIdx.x;
  const int lane = tid & 31;
  const int wid  = tid >> 5;
  const int nblk = blockIdx.x % 8;
  const int mblk = blockIdx.x / 8;
  const int mw = wid & 3, nw = wid >> 2;
  const int row = lane & 15, hi = lane >> 4;

  v8f acc[2][4] = {};
  gemm_mainloop_128x128(Ah + (size_t)mblk * 128 * DM, Wh + (size_t)nblk * 128 * DM,
                        &smA[0][0], &smB[0][0], acc, tid);

  const int mbase = mblk * 128 + mw * 32;
#pragma unroll
  for (int a = 0; a < 2; ++a) {
#pragma unroll
    for (int j = 0; j < 4; ++j) {
      const int n = nblk * 128 + nw * 64 + j * 16 + row;
#pragma unroll
      for (int r = 0; r < 8; ++r) {
        int m = mbase + a * 16 + r + hi * 8;
        out[(size_t)m * DM + n] = acc[a][j][r];
      }
    }
  }
}

extern "C" void kernel_launch(void* const* d_in, const int* in_sizes, int n_in,
                              void* d_out, int out_size, void* d_ws, size_t ws_size,
                              hipStream_t stream) {
  const float* X     = (const float*)d_in[0];   // [4,2048,1024]
  const float* Wqkv  = (const float*)d_in[1];   // [3072,1024]
  const float* Wproj = (const float*)d_in[2];   // [1024,1024]
  float* out = (float*)d_out;

  _Float16* ws = (_Float16*)d_ws;
  size_t o = 0;
  _Float16* Xh  = ws + o; o += (size_t)MTOT * DM;
  _Float16* Wqh = ws + o; o += (size_t)3 * DM * DM;
  _Float16* Wph = ws + o; o += (size_t)DM * DM;
  _Float16* Qd  = ws + o; o += (size_t)MTOT * DM;
  _Float16* Kd  = ws + o; o += (size_t)MTOT * DM;
  _Float16* Vt  = ws + o; o += (size_t)MTOT * DM;
  _Float16* Od  = ws + o; o += (size_t)MTOT * DM;

  const int nX = MTOT * DM, nWq = 3 * DM * DM, nWp = DM * DM;
  k_cvt_f32_f16<<<nX  / 1024, 256, 0, stream>>>(X, Xh, nX);
  k_cvt_f32_f16<<<nWq / 1024, 256, 0, stream>>>(Wqkv, Wqh, nWq);
  k_cvt_f32_f16<<<nWp / 1024, 256, 0, stream>>>(Wproj, Wph, nWp);

  // 64 M-blocks x 24 N-blocks, 128x128 macro-tile per 8-wave block
  k_gemm_qkv<<<64 * 24, 256, 0, stream>>>(Xh, Wqh, Qd, Kd, Vt);
  // 64 (b,h) x 16 query-blocks of 128
  k_attn<<<64 * 16, 256, 0, stream>>>(Qd, Kd, Vt, Od);
  // 64 M-blocks x 8 N-blocks
  k_gemm_proj<<<64 * 8, 256, 0, stream>>>(Od, Wph, out);
}
